// ExpertsFeedForward_52561809768547
// MI455X (gfx1250) — matrix operand order repaired and checked by reading, not compiled, not run on hardware
//
#include <hip/hip_runtime.h>
#include <stdint.h>

typedef _Float16 h16;
typedef __attribute__((ext_vector_type(16))) _Float16 v16h;
typedef __attribute__((ext_vector_type(8)))  _Float16 v8h;
typedef __attribute__((ext_vector_type(8)))  float    v8f;

#define NTOK 4096
#define DDIM 1024
#define HDIM 2048
#define NE 8
#define NE_FF 7

#define BM 64
#define BN 128
#define BK 64
#define LDA 72   // padded LDS row length in h16 elems (144 B)

// ---------------- workspace layout (bytes) ----------------
#define GATES_OFF  ((size_t)0)                            // float[4096][8]
#define COUNTS_OFF ((size_t)131072)                       // int[8]
#define OFFS_OFF   ((size_t)131104)                       // int[8]
#define LIST_OFF   ((size_t)131136)                       // int[7][4096]
#define SK16_OFF   ((size_t)262144)                       // h16[2048][1024]
#define SV16_OFF   (SK16_OFF + (size_t)HDIM * DDIM * 2)   // h16[1024][2048]
#define FK16_OFF   (SV16_OFF + (size_t)DDIM * HDIM * 2)   // h16[7][2048][1024]
#define FV16_OFF   (FK16_OFF + (size_t)NE_FF * HDIM * DDIM * 2)
#define HS_OFF     (FV16_OFF + (size_t)NE_FF * DDIM * HDIM * 2) // h16[4096][2048]
#define HE_OFF     (HS_OFF + (size_t)NTOK * HDIM * 2)     // h16[16448][2048] (64 slack rows)

__device__ __forceinline__ float gelu_f(float x) {
    const float c = 0.7978845608028654f;
    float t = tanhf(c * (x + 0.044715f * x * x * x));
    return 0.5f * x * (1.0f + t);
}

// async 16-byte global -> LDS copy (ASYNCcnt tracked)
__device__ __forceinline__ void async_cp16(const h16* g, const h16* l) {
    asm volatile("global_load_async_to_lds_b128 %0, %1, off"
                 :: "v"((uint32_t)(uintptr_t)l), "v"(g)
                 : "memory");
}
__device__ __forceinline__ void async_wait0() {
    asm volatile("s_wait_asynccnt 0x0" ::: "memory");
}

#define WMMA16(A, B, C) __builtin_amdgcn_wmma_f32_16x16x32_f16(false, A, false, B, (short)0, C, false, false)
#define SHUF16(LO, HI) __builtin_shufflevector(LO, HI, 0,1,2,3,4,5,6,7,8,9,10,11,12,13,14,15)

// ---------------- init ----------------
__global__ void init_kernel(int* counts, float* loss_out) {
    if (threadIdx.x < 8) counts[threadIdx.x] = 0;
    if (threadIdx.x == 8) loss_out[0] = 0.0f;
}

// ---------------- f32 -> f16 weight pre-conversion ----------------
__global__ void __launch_bounds__(256) cvt_kernel(const float* __restrict__ src,
                                                  h16* __restrict__ dst, int n) {
    int i = (blockIdx.x * 256 + threadIdx.x) * 8;
    if (i < n) {
        float4 a = *(const float4*)(src + i);
        float4 b = *(const float4*)(src + i + 4);
        v8h v;
        v[0] = (h16)a.x; v[1] = (h16)a.y; v[2] = (h16)a.z; v[3] = (h16)a.w;
        v[4] = (h16)b.x; v[5] = (h16)b.y; v[6] = (h16)b.z; v[7] = (h16)b.w;
        *(v8h*)(dst + i) = v;
    }
}

// ---------------- gating: one wave32 per token ----------------
__global__ void __launch_bounds__(256) gate_kernel(
    const float* __restrict__ x, const float* __restrict__ gw,
    const float* __restrict__ gb, float* __restrict__ gates,
    int* __restrict__ counts, int* __restrict__ lists)
{
    int lane = threadIdx.x & 31;
    int wave = threadIdx.x >> 5;
    int t = blockIdx.x * 8 + wave;
    if (t >= NTOK) return;

    float acc[NE];
    #pragma unroll
    for (int e = 0; e < NE; ++e) acc[e] = 0.0f;
    const float* xr = x + (size_t)t * DDIM;
    for (int d = lane; d < DDIM; d += 32) {
        float xv = xr[d];
        const float* gr = gw + (size_t)d * NE;
        #pragma unroll
        for (int e = 0; e < NE; ++e) acc[e] += xv * gr[e];
    }
    #pragma unroll
    for (int e = 0; e < NE; ++e) {
        #pragma unroll
        for (int off = 16; off > 0; off >>= 1)
            acc[e] += __shfl_xor(acc[e], off, 32);
    }
    if (lane == 0) {
        float p[NE];
        float mx = -1e30f;
        #pragma unroll
        for (int e = 0; e < NE; ++e) { p[e] = acc[e] + gb[e]; mx = fmaxf(mx, p[e]); }
        float s = 0.0f;
        #pragma unroll
        for (int e = 0; e < NE; ++e) { p[e] = __expf(p[e] - mx); s += p[e]; }
        float inv = 1.0f / s;
        #pragma unroll
        for (int e = 0; e < NE; ++e) p[e] *= inv;
        bool sel[NE];
        #pragma unroll
        for (int e = 0; e < NE; ++e) sel[e] = false;
        for (int k = 0; k < 4; ++k) {          // top-4, ties -> lowest index
            int best = -1; float bv = -1e30f;
            #pragma unroll
            for (int e = 0; e < NE; ++e)
                if (!sel[e] && p[e] > bv) { bv = p[e]; best = e; }
            sel[best] = true;
        }
        #pragma unroll
        for (int e = 0; e < NE; ++e) {
            float g = sel[e] ? p[e] : 0.0f;
            gates[(size_t)t * NE + e] = g;
            if (e < NE_FF && sel[e]) {
                int pos = atomicAdd(&counts[e], 1);
                lists[e * NTOK + pos] = t;
            }
        }
    }
}

// ---------------- scan ----------------
__global__ void scan_kernel(const int* counts, int* offs) {
    if (threadIdx.x == 0) {
        int a = 0;
        for (int e = 0; e < NE_FF; ++e) { offs[e] = a; a += counts[e]; }
        offs[NE_FF] = a;
    }
}

// ---------------- GEMM1: H = gelu(X @ K^T) -> f16 ----------------
// mode 0: shared (identity rows); mode 1: expert blockIdx.z, gathered rows
__global__ void __launch_bounds__(256) gemm1_kernel(
    const float* __restrict__ xin, const h16* __restrict__ keys16,
    h16* __restrict__ hout,
    const int* __restrict__ counts, const int* __restrict__ offs,
    const int* __restrict__ lists, int mode)
{
    __shared__ h16 As[2][BM * LDA];
    __shared__ h16 Bs[2][BN * LDA];

    int e = (mode == 0) ? 0 : blockIdx.z;
    int count = (mode == 0) ? NTOK : counts[e];
    int m_base = blockIdx.x * BM;
    if (m_base >= count) return;
    int n_base = blockIdx.y * BN;

    const h16* kbase = keys16 + (mode == 0 ? (size_t)0 : (size_t)e * HDIM * DDIM);
    int hrow_base = (mode == 0 ? 0 : offs[e]) + m_base;

    int tid = threadIdx.x;
    int sr = tid >> 2, sq = tid & 3;           // A staging: row, 16-elem quad
    int aslot = m_base + sr;
    int atok = (mode == 0) ? aslot : ((aslot < count) ? lists[e * NTOK + aslot] : -1);
    const float* arow = (atok >= 0) ? (xin + (size_t)atok * DDIM) : nullptr;

    int lane = tid & 31, wave = tid >> 5;
    int wm = wave >> 2, wn = wave & 3;          // 2 x 4 wave grid
    int fr = lane & 15;
    int kb0 = (lane < 16) ? 0 : 16;             // A: contiguous 16 K pattern
    int kc0 = (lane < 16) ? 0 : 16;             // B: contiguous 16 K pattern

    v8f c00 = {}, c01 = {}, c10 = {}, c11 = {};

    auto stage = [&](int kb, h16* Asb, h16* Bsb) {
        #pragma unroll
        for (int i = 0; i < 4; ++i) {
            int ch = tid * 4 + i;
            int row = ch >> 3, sub = ch & 7;
            async_cp16(kbase + (size_t)(n_base + row) * DDIM + kb + sub * 8,
                       &Bsb[row * LDA + sub * 8]);
        }
        v8h u0 = {}, u1 = {};
        if (arow) {
            const float4* p = (const float4*)(arow + kb + sq * 16);
            float4 f0 = p[0], f1 = p[1], f2 = p[2], f3 = p[3];
            u0[0] = (h16)f0.x; u0[1] = (h16)f0.y; u0[2] = (h16)f0.z; u0[3] = (h16)f0.w;
            u0[4] = (h16)f1.x; u0[5] = (h16)f1.y; u0[6] = (h16)f1.z; u0[7] = (h16)f1.w;
            u1[0] = (h16)f2.x; u1[1] = (h16)f2.y; u1[2] = (h16)f2.z; u1[3] = (h16)f2.w;
            u1[4] = (h16)f3.x; u1[5] = (h16)f3.y; u1[6] = (h16)f3.z; u1[7] = (h16)f3.w;
            if (kb + BK < DDIM) __builtin_prefetch(arow + kb + BK + sq * 16, 0, 1);
        }
        *(v8h*)&Asb[sr * LDA + sq * 16]     = u0;
        *(v8h*)&Asb[sr * LDA + sq * 16 + 8] = u1;
    };
    auto mma = [&](const h16* Asb, const h16* Bsb) {
        #pragma unroll
        for (int s = 0; s < 2; ++s) {
            int ks = s * 32;
            int ai = (wm * 32 + fr) * LDA + ks + kb0;
            v16h a0 = SHUF16(*(const v8h*)&Asb[ai], *(const v8h*)&Asb[ai + 8]);
            int aj = ai + 16 * LDA;
            v16h a1 = SHUF16(*(const v8h*)&Asb[aj], *(const v8h*)&Asb[aj + 8]);
            int bi = (wn * 32 + fr) * LDA + ks + kc0;
            v16h b0 = SHUF16(*(const v8h*)&Bsb[bi], *(const v8h*)&Bsb[bi + 8]);
            int bj = bi + 16 * LDA;
            v16h b1 = SHUF16(*(const v8h*)&Bsb[bj], *(const v8h*)&Bsb[bj + 8]);
            c00 = WMMA16(a0, b0, c00);
            c01 = WMMA16(a0, b1, c01);
            c10 = WMMA16(a1, b0, c10);
            c11 = WMMA16(a1, b1, c11);
        }
    };

    // two-deep LDS ping-pong pipeline
    stage(0, As[0], Bs[0]);
    async_wait0();
    __syncthreads();
    int cur = 0;
    for (int kb = 0; kb < DDIM; kb += BK) {
        int nxt = cur ^ 1;
        if (kb + BK < DDIM) stage(kb + BK, As[nxt], Bs[nxt]);
        mma(As[cur], Bs[cur]);
        async_wait0();
        __syncthreads();
        cur = nxt;
    }

    int rb = (lane < 16) ? 0 : 8;
    int col0 = n_base + wn * 32 + fr;
    #pragma unroll
    for (int j = 0; j < 8; ++j) {
        int m0 = wm * 32 + rb + j;
        int m1 = m0 + 16;
        if (m_base + m0 < count) {
            size_t r = (size_t)(hrow_base + m0) * HDIM;
            hout[r + col0]      = (h16)gelu_f(c00[j]);
            hout[r + col0 + 16] = (h16)gelu_f(c01[j]);
        }
        if (m_base + m1 < count) {
            size_t r = (size_t)(hrow_base + m1) * HDIM;
            hout[r + col0]      = (h16)gelu_f(c10[j]);
            hout[r + col0 + 16] = (h16)gelu_f(c11[j]);
        }
    }
}

// ---------------- GEMM2: Y = Hact @ V^T (+const / gate-scaled atomic) ----------------
__global__ void __launch_bounds__(256) gemm2_kernel(
    const h16* __restrict__ hact, const h16* __restrict__ values16,
    const float* __restrict__ gates, const float* __restrict__ cjump,
    float* __restrict__ out,
    const int* __restrict__ counts, const int* __restrict__ offs,
    const int* __restrict__ lists, int mode)
{
    __shared__ h16 As[2][BM * LDA];
    __shared__ h16 Bs[2][BN * LDA];

    int e = (mode == 0) ? 0 : blockIdx.z;
    int count = (mode == 0) ? NTOK : counts[e];
    int m_base = blockIdx.x * BM;
    if (m_base >= count) return;
    int n_base = blockIdx.y * BN;

    const h16* vbase = values16 + (mode == 0 ? (size_t)0 : (size_t)e * DDIM * HDIM);
    const h16* abase = hact + (mode == 0 ? (size_t)0 : (size_t)offs[e] * HDIM);

    int tid = threadIdx.x;
    int lane = tid & 31, wave = tid >> 5;
    int wm = wave >> 2, wn = wave & 3;
    int fr = lane & 15;
    int kb0 = (lane < 16) ? 0 : 16;
    int kc0 = (lane < 16) ? 0 : 16;

    v8f c00 = {}, c01 = {}, c10 = {}, c11 = {};

    auto stage = [&](int kb, h16* Asb, h16* Bsb) {
        #pragma unroll
        for (int i = 0; i < 2; ++i) {
            int ch = tid * 2 + i;
            int row = ch >> 3, sub = ch & 7;
            async_cp16(abase + (size_t)(m_base + row) * HDIM + kb + sub * 8,
                       &Asb[row * LDA + sub * 8]);
        }
        #pragma unroll
        for (int i = 0; i < 4; ++i) {
            int ch = tid * 4 + i;
            int row = ch >> 3, sub = ch & 7;
            async_cp16(vbase + (size_t)(n_base + row) * HDIM + kb + sub * 8,
                       &Bsb[row * LDA + sub * 8]);
        }
    };
    auto mma = [&](const h16* Asb, const h16* Bsb) {
        #pragma unroll
        for (int s = 0; s < 2; ++s) {
            int ks = s * 32;
            int ai = (wm * 32 + fr) * LDA + ks + kb0;
            v16h a0 = SHUF16(*(const v8h*)&Asb[ai], *(const v8h*)&Asb[ai + 8]);
            int aj = ai + 16 * LDA;
            v16h a1 = SHUF16(*(const v8h*)&Asb[aj], *(const v8h*)&Asb[aj + 8]);
            int bi = (wn * 32 + fr) * LDA + ks + kc0;
            v16h b0 = SHUF16(*(const v8h*)&Bsb[bi], *(const v8h*)&Bsb[bi + 8]);
            int bj = bi + 16 * LDA;
            v16h b1 = SHUF16(*(const v8h*)&Bsb[bj], *(const v8h*)&Bsb[bj + 8]);
            c00 = WMMA16(a0, b0, c00);
            c01 = WMMA16(a0, b1, c01);
            c10 = WMMA16(a1, b0, c10);
            c11 = WMMA16(a1, b1, c11);
        }
    };

    // two-deep LDS ping-pong pipeline
    stage(0, As[0], Bs[0]);
    async_wait0();
    __syncthreads();
    int cur = 0;
    for (int kb = 0; kb < HDIM; kb += BK) {
        int nxt = cur ^ 1;
        if (kb + BK < HDIM) stage(kb + BK, As[nxt], Bs[nxt]);
        mma(As[cur], Bs[cur]);
        async_wait0();
        __syncthreads();
        cur = nxt;
    }

    int rb = (lane < 16) ? 0 : 8;
    int col0 = n_base + wn * 32 + fr;
    #pragma unroll
    for (int i = 0; i < 2; ++i) {
        v8f* cl = (i == 0) ? &c00 : &c10;
        v8f* cr = (i == 0) ? &c01 : &c11;
        #pragma unroll
        for (int j = 0; j < 8; ++j) {
            int mrow = wm * 32 + i * 16 + rb + j;
            int slot = m_base + mrow;
            if (mode == 0) {
                int token = slot;
                float g7 = gates[(size_t)token * NE + 7];
                size_t o = (size_t)token * DDIM;
                out[o + col0]      = (*cl)[j] + g7 * cjump[col0];
                out[o + col0 + 16] = (*cr)[j] + g7 * cjump[col0 + 16];
            } else if (slot < count) {
                int token = lists[e * NTOK + slot];
                float g = gates[(size_t)token * NE + e];
                size_t o = (size_t)token * DDIM;
                unsafeAtomicAdd(&out[o + col0],      g * (*cl)[j]);
                unsafeAtomicAdd(&out[o + col0 + 16], g * (*cr)[j]);
            }
        }
    }
}

extern "C" void kernel_launch(void* const* d_in, const int* in_sizes, int n_in,
                              void* d_out, int out_size, void* d_ws, size_t ws_size,
                              hipStream_t stream) {
    const float* x  = (const float*)d_in[0];
    const float* gw = (const float*)d_in[1];
    const float* gb = (const float*)d_in[2];
    const float* fk = (const float*)d_in[3];
    const float* fv = (const float*)d_in[4];
    const float* sk = (const float*)d_in[5];
    const float* sv = (const float*)d_in[6];
    const float* cj = (const float*)d_in[7];
    float* out = (float*)d_out;

    char* ws = (char*)d_ws;
    float* gates = (float*)(ws + GATES_OFF);
    int* counts  = (int*)(ws + COUNTS_OFF);
    int* offs    = (int*)(ws + OFFS_OFF);
    int* lists   = (int*)(ws + LIST_OFF);
    h16* SK16    = (h16*)(ws + SK16_OFF);
    h16* SV16    = (h16*)(ws + SV16_OFF);
    h16* FK16    = (h16*)(ws + FK16_OFF);
    h16* FV16    = (h16*)(ws + FV16_OFF);
    h16* HS      = (h16*)(ws + HS_OFF);
    h16* HE      = (h16*)(ws + HE_OFF);

    init_kernel<<<dim3(1), dim3(32), 0, stream>>>(counts, out + (size_t)NTOK * DDIM);
    gate_kernel<<<dim3(NTOK / 8), dim3(256), 0, stream>>>(x, gw, gb, gates, counts, lists);
    scan_kernel<<<dim3(1), dim3(32), 0, stream>>>(counts, offs);

    // weight pre-conversion to f16
    int nSK = HDIM * DDIM, nFK = NE_FF * HDIM * DDIM;
    cvt_kernel<<<dim3(nSK / 2048), dim3(256), 0, stream>>>(sk, SK16, nSK);
    cvt_kernel<<<dim3(nSK / 2048), dim3(256), 0, stream>>>(sv, SV16, nSK);
    cvt_kernel<<<dim3(nFK / 2048), dim3(256), 0, stream>>>(fk, FK16, nFK);
    cvt_kernel<<<dim3(nFK / 2048), dim3(256), 0, stream>>>(fv, FV16, nFK);

    // GEMM1: H = gelu(X @ K^T)   (N = 2048 -> 16 N-tiles of 128)
    gemm1_kernel<<<dim3(64, 16, 1),     dim3(256), 0, stream>>>(x, SK16, HS, counts, offs, lists, 0);
    gemm1_kernel<<<dim3(64, 16, NE_FF), dim3(256), 0, stream>>>(x, FK16, HE, counts, offs, lists, 1);
    // GEMM2: Y = H @ V^T         (N = 1024 -> 8 N-tiles of 128)
    gemm2_kernel<<<dim3(64, 8, 1),      dim3(256), 0, stream>>>(HS, SV16, gates, cj, out, counts, offs, lists, 0);
    gemm2_kernel<<<dim3(64, 8, NE_FF),  dim3(256), 0, stream>>>(HE, FV16, gates, cj, out, counts, offs, lists, 1);
}
